// DeepfakeDetector_46282567582258
// MI455X (gfx1250) — matrix operand order repaired and checked by reading, not compiled, not run on hardware
//
#include <hip/hip_runtime.h>
#include <hip/hip_bf16.h>

typedef __attribute__((ext_vector_type(16))) _Float16 v16h;
typedef __attribute__((ext_vector_type(8)))  float    v8f;

// ---------------------------------------------------------------------------
// Preprocess: build 16-channel f16 input (B,16,224,224):
//   ch 0-2  : image RGB
//   ch 3-14 : Haar DWT (cA,cH,cV,cD x RGB) bilinearly upsampled 112->224
//   ch 15   : Sobel magnitude of gray
// ---------------------------------------------------------------------------
__global__ void prep_k(const float* __restrict__ img, _Float16* __restrict__ x, int Bn) {
    const int H = 224, W = 224;
    long long i = (long long)blockIdx.x * blockDim.x + threadIdx.x;
    long long tot = (long long)Bn * H * W;
    if (i >= tot) return;
    int xp = (int)(i % W);
    long long t = i / W;
    int y = (int)(t % H);
    int b = (int)(t / H);
    const long long ib = (long long)b * 3 * H * W;
    const long long ob = (long long)b * 16 * H * W + (long long)y * W + xp;
    const long long CS = (long long)H * W;

    float r = img[ib + 0 * CS + (long long)y * W + xp];
    float g = img[ib + 1 * CS + (long long)y * W + xp];
    float bl = img[ib + 2 * CS + (long long)y * W + xp];
    x[ob + 0 * CS] = (_Float16)r;
    x[ob + 1 * CS] = (_Float16)g;
    x[ob + 2 * CS] = (_Float16)bl;

    float sy = (y + 0.5f) * 0.5f - 0.5f;
    float sx = (xp + 0.5f) * 0.5f - 0.5f;
    int iy0 = (int)floorf(sy), ix0 = (int)floorf(sx);
    float fy = sy - (float)iy0, fx = sx - (float)ix0;
    float acc[12];
#pragma unroll
    for (int k = 0; k < 12; ++k) acc[k] = 0.f;
    for (int dy = 0; dy < 2; ++dy) {
        for (int dx = 0; dx < 2; ++dx) {
            int yy = min(max(iy0 + dy, 0), 111);
            int xx = min(max(ix0 + dx, 0), 111);
            float wgt = (dy ? fy : 1.f - fy) * (dx ? fx : 1.f - fx);
            if (wgt == 0.f) continue;
#pragma unroll
            for (int c = 0; c < 3; ++c) {
                const float* p = img + ib + (long long)c * CS;
                float a  = p[(2 * yy) * W + 2 * xx];
                float bb = p[(2 * yy) * W + 2 * xx + 1];
                float cc = p[(2 * yy + 1) * W + 2 * xx];
                float dd = p[(2 * yy + 1) * W + 2 * xx + 1];
                acc[0 + c] += wgt * 0.5f * (a + bb + cc + dd);
                acc[3 + c] += wgt * 0.5f * (a + bb - cc - dd);
                acc[6 + c] += wgt * 0.5f * (a - bb + cc - dd);
                acc[9 + c] += wgt * 0.5f * (a - bb - cc + dd);
            }
        }
    }
#pragma unroll
    for (int c = 0; c < 12; ++c) x[ob + (long long)(3 + c) * CS] = (_Float16)acc[c];

    const float kxw[9] = {-1, 0, 1, -2, 0, 2, -1, 0, 1};
    const float kyw[9] = {-1, -2, -1, 0, 0, 0, 1, 2, 1};
    float gx = 0.f, gy = 0.f;
#pragma unroll
    for (int dy = -1; dy <= 1; ++dy) {
#pragma unroll
        for (int dx = -1; dx <= 1; ++dx) {
            int yy = y + dy, xx = xp + dx;
            float gv = 0.f;
            if ((unsigned)yy < (unsigned)H && (unsigned)xx < (unsigned)W) {
                gv = (img[ib + (long long)yy * W + xx] +
                      img[ib + CS + (long long)yy * W + xx] +
                      img[ib + 2 * CS + (long long)yy * W + xx]) * (1.f / 3.f);
            }
            int ki = (dy + 1) * 3 + (dx + 1);
            gx += kxw[ki] * gv;
            gy += kyw[ki] * gv;
        }
    }
    x[ob + 15 * CS] = (_Float16)sqrtf(gx * gx + gy * gy);
}

// ---------------------------------------------------------------------------
// Fold BN: sb[0..C-1]=scale, sb[C..2C-1]=bias
// ---------------------------------------------------------------------------
__global__ void bn_fold_k(const float* g, const float* be, const float* mn,
                          const float* vr, float* sb, int C) {
    int i = blockIdx.x * blockDim.x + threadIdx.x;
    if (i < C) {
        float s = g[i] * rsqrtf(vr[i] + 1e-5f);
        sb[i] = s;
        sb[C + i] = be[i] - mn[i] * s;
    }
}

// ---------------------------------------------------------------------------
// Pack conv weights (Cout, Cin*KH*KW) fp32 into CDNA5 16x32 f16 A-fragment
// lane layout.  Packed element index = ((mt*KB+kb)*32 + lane)*16 + h.
// ---------------------------------------------------------------------------
__global__ void pack_w_k(const float* __restrict__ w, _Float16* __restrict__ p,
                         int Cout, int Kdim, int KB) {
    long long idx = (long long)blockIdx.x * blockDim.x + threadIdx.x;
    long long total = (long long)(Cout / 16) * KB * 512;
    if (idx >= total) return;
    int h = (int)(idx & 15);
    int lane = (int)((idx >> 4) & 31);
    long long tl = idx >> 9;
    int kb = (int)(tl % KB);
    int mt = (int)(tl / KB);
    int m_local = (lane < 16) ? lane : lane - 16;
    int k_local = (lane < 16) ? ((h < 8) ? h : 16 + (h - 8))
                              : ((h < 8) ? 8 + h : 24 + (h - 8));
    int m = mt * 16 + m_local;
    int k = kb * 32 + k_local;
    float val = (k < Kdim) ? w[(long long)m * Kdim + k] : 0.f;
    p[idx] = (_Float16)val;
}

// ---------------------------------------------------------------------------
// Implicit-GEMM conv + folded BN + optional residual + optional ReLU.
// Templated on TKK=KH*KW, TKW=KW (constant-divisor k-decode) and KCHK
// (whether Kdim may be a non-multiple of 32; only the 7x7 stem needs it).
// Each wave computes a 64(Cout) x 32(pixels) tile: 4 A fragments x
// 2 B fragments -> EIGHT v_wmma_f32_16x16x32_f16 per K-step.  The gather
// (32 predicated u16 loads) and the k-decode are shared across all eight,
// so per-WMMA gather cost is 4 loads.  A-fragment loads are 1 aligned
// 32-byte load per tile per K-step.
// ---------------------------------------------------------------------------
template <int TKK, int TKW, bool KCHK>
__global__ __launch_bounds__(256) void conv_wmma_k(
    const _Float16* __restrict__ act, const _Float16* __restrict__ wp,
    const float* __restrict__ scale, const float* __restrict__ bias,
    const _Float16* __restrict__ res, _Float16* __restrict__ out,
    int Bn, int Cin, int Hin, int Win, int Cout, int Hout, int Wout,
    int stride, int pad, int relu) {
    const int lane = threadIdx.x & 31;
    const int wave = threadIdx.x >> 5;
    const int Ntot = Bn * Hout * Wout;

    const int tilebase = ((int)blockIdx.x * 8 + wave) * 32;
    int n0 = tilebase + (lane & 15);
    int n1 = n0 + 16;
    const bool nv0 = n0 < Ntot;
    const bool nv1 = n1 < Ntot;
    int nc0 = nv0 ? n0 : 0;
    int nc1 = nv1 ? n1 : 0;
    int ox0 = nc0 % Wout, t0 = nc0 / Wout, oy0 = t0 % Hout, b0 = t0 / Hout;
    int ox1 = nc1 % Wout, t1 = nc1 / Wout, oy1 = t1 % Hout, b1 = t1 / Hout;

    const int Kdim = Cin * TKK;
    const int KB = (Kdim + 31) / 32;
    const int khi = (lane < 16) ? 0 : 8;
    const int iyb0 = oy0 * stride - pad, ixb0 = ox0 * stride - pad;
    const int iyb1 = oy1 * stride - pad, ixb1 = ox1 * stride - pad;
    const long long plane = (long long)Hin * Win;
    const long long actb0 = (long long)b0 * Cin * plane;
    const long long actb1 = (long long)b1 * Cin * plane;
    // four M-tiles (16 rows each): packed tile indices 4*blockIdx.y .. +3
    const _Float16* wbase = wp + ((long long)blockIdx.y * 4) * KB * 512 + (long long)lane * 16;
    const long long tstep = (long long)KB * 512;

    v8f c0[4], c1[4];
#pragma unroll
    for (int at = 0; at < 4; ++at) {
        c0[at] = (v8f){0.f, 0.f, 0.f, 0.f, 0.f, 0.f, 0.f, 0.f};
        c1[at] = c0[at];
    }

    for (int kb = 0; kb < KB; ++kb) {
        v16h a[4];
#pragma unroll
        for (int at = 0; at < 4; ++at) {
            a[at] = *(const v16h*)(wbase + at * tstep + (long long)kb * 512);
            if (kb + 1 < KB)
                __builtin_prefetch(
                    (const void*)(wbase + at * tstep + (long long)(kb + 1) * 512), 0, 0);
        }

        union { v16h v; _Float16 h[16]; } bu0, bu1;
        const int k0 = kb * 32 + khi;
#pragma unroll
        for (int e = 0; e < 16; ++e) {
            int k = k0 + ((e < 8) ? e : 8 + e);
            _Float16 v0 = (_Float16)0.f;
            _Float16 v1 = (_Float16)0.f;
            if (!KCHK || k < Kdim) {
                // constant-divisor decode (TKK, TKW are template constants)
                int cin = k / TKK;
                int rr = k - cin * TKK;
                int ky = rr / TKW;
                int kx = rr - ky * TKW;
                long long cb = (long long)cin * plane;
                int iy0 = iyb0 + ky, ix0 = ixb0 + kx;
                int iy1 = iyb1 + ky, ix1 = ixb1 + kx;
                if (nv0 && (unsigned)iy0 < (unsigned)Hin && (unsigned)ix0 < (unsigned)Win)
                    v0 = act[actb0 + cb + (long long)iy0 * Win + ix0];
                if (nv1 && (unsigned)iy1 < (unsigned)Hin && (unsigned)ix1 < (unsigned)Win)
                    v1 = act[actb1 + cb + (long long)iy1 * Win + ix1];
            }
            bu0.h[e] = v0;
            bu1.h[e] = v1;
        }
        // EXEC is all-1s here (no divergent branch encloses these calls).
        // Eight independent accumulators: no D->A/B RAW hazard between them.
#pragma unroll
        for (int at = 0; at < 4; ++at) {
            c0[at] = __builtin_amdgcn_wmma_f32_16x16x32_f16(
                false, a[at], false, bu0.v, (short)0, c0[at], false, false);
            c1[at] = __builtin_amdgcn_wmma_f32_16x16x32_f16(
                false, a[at], false, bu1.v, (short)0, c1[at], false, false);
        }
    }

    // Epilogue: VGPR r holds M=r (lanes 0-15) / M=r+8 (lanes 16-31), N=lane%16
    const int mbase = (int)blockIdx.y * 64 + ((lane < 16) ? 0 : 8);
#pragma unroll
    for (int at = 0; at < 4; ++at) {
#pragma unroll
        for (int r = 0; r < 8; ++r) {
            int m = mbase + at * 16 + r;
            float sc = scale[m], bi = bias[m];
            float v0 = c0[at][r] * sc + bi;
            float v1 = c1[at][r] * sc + bi;
            long long o0 = (((long long)b0 * Cout + m) * Hout + oy0) * Wout + ox0;
            long long o1 = (((long long)b1 * Cout + m) * Hout + oy1) * Wout + ox1;
            if (res) {
                v0 += (float)res[o0];
                v1 += (float)res[o1];
            }
            if (relu) {
                v0 = fmaxf(v0, 0.f);
                v1 = fmaxf(v1, 0.f);
            }
            if (nv0) out[o0] = (_Float16)v0;
            if (nv1) out[o1] = (_Float16)v1;
        }
    }
}

// ---------------------------------------------------------------------------
// MaxPool 3x3 stride 2 pad 1 (f16 -> f16)
// ---------------------------------------------------------------------------
__global__ void maxpool_k(const _Float16* __restrict__ in, _Float16* __restrict__ out,
                          int Bn, int C, int Hin, int Win, int Hout, int Wout) {
    long long i = (long long)blockIdx.x * blockDim.x + threadIdx.x;
    long long tot = (long long)Bn * C * Hout * Wout;
    if (i >= tot) return;
    int ox = (int)(i % Wout);
    long long t = i / Wout;
    int oy = (int)(t % Hout);
    t /= Hout;
    int ch = (int)(t % C);
    int b = (int)(t / C);
    float m = -3.0e38f;
    for (int ky = 0; ky < 3; ++ky) {
        int iy = oy * 2 - 1 + ky;
        if ((unsigned)iy >= (unsigned)Hin) continue;
        for (int kx = 0; kx < 3; ++kx) {
            int ix = ox * 2 - 1 + kx;
            if ((unsigned)ix >= (unsigned)Win) continue;
            float v = (float)in[(((long long)b * C + ch) * Hin + iy) * Win + ix];
            m = fmaxf(m, v);
        }
    }
    out[i] = (_Float16)m;
}

// ---------------------------------------------------------------------------
// Global average pool (f16 -> f32)
// ---------------------------------------------------------------------------
__global__ void avgpool_k(const _Float16* __restrict__ in, float* __restrict__ feat,
                          int Bn, int C, int HW) {
    int i = blockIdx.x * blockDim.x + threadIdx.x;
    if (i >= Bn * C) return;
    const _Float16* p = in + (long long)i * HW;
    float s = 0.f;
    for (int j = 0; j < HW; ++j) s += (float)p[j];
    feat[i] = s / (float)HW;
}

// ---------------------------------------------------------------------------
// FC:  out[b,j] = act(sum_k feat[b,k]*w[k,j] + bias[j])   (fp32, tiny)
// ---------------------------------------------------------------------------
__global__ void fc_k(const float* __restrict__ feat, const float* __restrict__ w,
                     const float* __restrict__ bias, float* __restrict__ out,
                     int Bn, int K, int N, int relu) {
    int i = blockIdx.x * blockDim.x + threadIdx.x;
    if (i >= Bn * N) return;
    int j = i % N, b = i / N;
    float s = bias[j];
    for (int k = 0; k < K; ++k) s += feat[(long long)b * K + k] * w[(long long)k * N + j];
    if (relu) s = fmaxf(s, 0.f);
    out[i] = s;
}

// ---------------------------------------------------------------------------
// Host orchestration
// ---------------------------------------------------------------------------
// Input index map (depth-first insertion order of setup_inputs):
//  0: image (32,3,224,224)
//  1: conv1 w (64,16,7,7)   2-5: bn1 g/b/m/v
//  per block: conv1 w, bn1 x4, conv2 w, bn2 x4 [, down_conv w, down_bn x4]
//  101: fc1_w (512,256) 102: fc1_b 103: fc2_w (256,2) 104: fc2_b
struct Blk { int w1, bn1, w2, bn2, wd, bnd, cin, cout, stride; };
static const Blk g_blks[8] = {
    { 6,  7, 11, 12, -1, -1,  64,  64, 1},
    {16, 17, 21, 22, -1, -1,  64,  64, 1},
    {26, 27, 31, 32, 36, 37,  64, 128, 2},
    {41, 42, 46, 47, -1, -1, 128, 128, 1},
    {51, 52, 56, 57, 61, 62, 128, 256, 2},
    {66, 67, 71, 72, -1, -1, 256, 256, 1},
    {76, 77, 81, 82, 86, 87, 256, 512, 2},
    {91, 92, 96, 97, -1, -1, 512, 512, 1},
};

static void* ws_take(char* ws, size_t& off, size_t bytes) {
    void* p = ws + off;
    off = (off + bytes + 255) & ~(size_t)255;
    return p;
}

extern "C" void kernel_launch(void* const* d_in, const int* in_sizes, int n_in,
                              void* d_out, int out_size, void* d_ws, size_t ws_size,
                              hipStream_t stream) {
    (void)in_sizes; (void)n_in; (void)out_size; (void)ws_size;
    const int Bn = 32;
    const float* image = (const float*)d_in[0];
    char* ws = (char*)d_ws;
    size_t off = 0;

    const size_t ACT_HALVES = (size_t)Bn * 16 * 224 * 224;  // == 32*64*112*112
    _Float16* buf[4];
    for (int i = 0; i < 4; ++i) buf[i] = (_Float16*)ws_take(ws, off, ACT_HALVES * 2);
    float* feat = (float*)ws_take(ws, off, (size_t)Bn * 512 * 4);
    float* hbuf = (float*)ws_take(ws, off, (size_t)Bn * 256 * 4);

    auto fold = [&](int gi, int C) -> float* {
        float* sb = (float*)ws_take(ws, off, (size_t)C * 2 * 4);
        bn_fold_k<<<(C + 63) / 64, 64, 0, stream>>>(
            (const float*)d_in[gi], (const float*)d_in[gi + 1],
            (const float*)d_in[gi + 2], (const float*)d_in[gi + 3], sb, C);
        return sb;
    };
    auto pack = [&](int wi, int Cout, int Cin, int KH, int KW) -> _Float16* {
        int Kdim = Cin * KH * KW;
        int KB = (Kdim + 31) / 32;
        size_t halves = (size_t)(Cout / 16) * KB * 512;
        _Float16* p = (_Float16*)ws_take(ws, off, halves * 2);
        long long tot = (long long)halves;
        pack_w_k<<<(unsigned)((tot + 255) / 256), 256, 0, stream>>>(
            (const float*)d_in[wi], p, Cout, Kdim, KB);
        return p;
    };
    auto conv = [&](const _Float16* a, _Float16* o, const _Float16* r,
                    int wi, int bni, int Cin, int Hin, int Win,
                    int Cout, int Hout, int Wout, int KH, int KW,
                    int s, int pd, int relu) {
        float* sb = fold(bni, Cout);
        _Float16* wp = pack(wi, Cout, Cin, KH, KW);
        long long Ntot = (long long)Bn * Hout * Wout;
        dim3 grid((unsigned)((Ntot + 255) / 256), (unsigned)(Cout / 64));
        if (KH == 3) {
            conv_wmma_k<9, 3, false><<<grid, 256, 0, stream>>>(
                a, wp, sb, sb + Cout, r, o, Bn, Cin, Hin, Win, Cout, Hout, Wout,
                s, pd, relu);
        } else if (KH == 1) {
            conv_wmma_k<1, 1, false><<<grid, 256, 0, stream>>>(
                a, wp, sb, sb + Cout, r, o, Bn, Cin, Hin, Win, Cout, Hout, Wout,
                s, pd, relu);
        } else {  // 7x7 stem (Kdim=784, not a multiple of 32 -> keep k<Kdim guard)
            conv_wmma_k<49, 7, true><<<grid, 256, 0, stream>>>(
                a, wp, sb, sb + Cout, r, o, Bn, Cin, Hin, Win, Cout, Hout, Wout,
                s, pd, relu);
        }
    };

    // 1) preprocess -> buf[0] (32,16,224,224)
    {
        long long tot = (long long)Bn * 224 * 224;
        prep_k<<<(unsigned)((tot + 255) / 256), 256, 0, stream>>>(image, buf[0], Bn);
    }
    // 2) stem conv 7x7 s2 p3 + bn + relu -> buf[1] (32,64,112,112)
    conv(buf[0], buf[1], nullptr, 1, 2, 16, 224, 224, 64, 112, 112, 7, 7, 2, 3, 1);
    // 3) maxpool -> buf[2] (32,64,56,56)
    {
        long long tot = (long long)Bn * 64 * 56 * 56;
        maxpool_k<<<(unsigned)((tot + 255) / 256), 256, 0, stream>>>(
            buf[1], buf[2], Bn, 64, 112, 112, 56, 56);
    }
    // 4) residual blocks
    int cur = 2, Hc = 56;
    for (int t = 0; t < 8; ++t) {
        const Blk& B = g_blks[t];
        int Ho = (B.stride == 2) ? Hc / 2 : Hc;
        int i1 = (cur + 1) & 3, i2 = (cur + 2) & 3, i3 = (cur + 3) & 3;
        conv(buf[cur], buf[i1], nullptr, B.w1, B.bn1,
             B.cin, Hc, Hc, B.cout, Ho, Ho, 3, 3, B.stride, 1, 1);
        const _Float16* idn = buf[cur];
        if (B.wd >= 0) {
            conv(buf[cur], buf[i2], nullptr, B.wd, B.bnd,
                 B.cin, Hc, Hc, B.cout, Ho, Ho, 1, 1, B.stride, 0, 0);
            idn = buf[i2];
        }
        conv(buf[i1], buf[i3], idn, B.w2, B.bn2,
             B.cout, Ho, Ho, B.cout, Ho, Ho, 3, 3, 1, 1, 1);
        cur = i3;
        Hc = Ho;
    }
    // 5) global average pool -> feat (32,512) fp32
    avgpool_k<<<(Bn * 512 + 255) / 256, 256, 0, stream>>>(buf[cur], feat, Bn, 512, Hc * Hc);
    // 6) FC1 + relu -> hbuf, FC2 -> d_out
    fc_k<<<(Bn * 256 + 255) / 256, 256, 0, stream>>>(
        feat, (const float*)d_in[101], (const float*)d_in[102], hbuf, Bn, 512, 256, 1);
    fc_k<<<(Bn * 2 + 63) / 64, 64, 0, stream>>>(
        hbuf, (const float*)d_in[103], (const float*)d_in[104], (float*)d_out, Bn, 256, 2, 0);
}
